// Next_Node_Probability_Calculator_for_group_28097676051152
// MI455X (gfx1250) — matrix operand order repaired and checked by reading, not compiled, not run on hardware
//
#include <hip/hip_runtime.h>
#include <hip/hip_bf16.h>

// MI455X (gfx1250) fused pointer-attention.
// Roofline: ~150MB traffic (mask read + probs write dominate) => ~7us @ 23.3TB/s;
// ~17 GFLOP of matrix math on v_wmma_f32_16x16x32_f16 sits well under that floor.
// Round 2: all WMMA B-operands (weights, V) and the Q A-operand are pre-packed in
// exact operand lane layout so hot-loop loads are single 32B v16h vector loads.

typedef __attribute__((ext_vector_type(16))) _Float16 v16h;
typedef __attribute__((ext_vector_type(4)))  _Float16 v4h;
typedef __attribute__((ext_vector_type(8)))  float    v8f;

constexpr int Bn = 64;
constexpr int Nn = 512;
constexpr int Gn = 512;
constexpr int En = 128;   // embedding dim == H*D
constexpr int Dn = 16;    // head dim

// ---- WMMA operand layout helpers (CDNA5 ISA §7.12.2) ----
// A-matrix 16x32 f16: lane r=lane&15 holds row r; half h -> K index:
__device__ __forceinline__ int a_kidx(int h, int hi) {
  return (hi << 4) + h;
}

// A tile gather from f16 LDS/global (row-major, stride ld)
__device__ __forceinline__ v16h load_a_f16(const _Float16* src, int ld, int lane) {
  int r = lane & 15, hi = lane >> 4;
  v16h a;
#pragma unroll
  for (int h = 0; h < 16; ++h) a[h] = src[r * ld + a_kidx(h, hi)];
  return a;
}

// A tile gather from f32 LDS (row-major, stride ld), converting
__device__ __forceinline__ v16h load_a_f32(const float* src, int ld, int lane) {
  int r = lane & 15, hi = lane >> 4;
  v16h a;
#pragma unroll
  for (int h = 0; h < 16; ++h) a[h] = (_Float16)src[r * ld + a_kidx(h, hi)];
  return a;
}

// Packed operand tile: [tile][lane][16 halfs] -> one aligned 32B load
__device__ __forceinline__ v16h load_packed(const _Float16* p, int tile, int lane) {
  return *(const v16h*)(p + ((size_t)tile << 9) + (lane << 4));
}

__device__ __forceinline__ v8f wmma32(v16h a, v16h b, v8f c) {
  return __builtin_amdgcn_wmma_f32_16x16x32_f16(false, a, false, b, (short)0, c,
                                                false, false);
}

// -------- Kernel 0: pack 128x128 f32 weights into f16 B-operand tiles --------
// tile t = kc*8 + c0 (kc: 32-row chunk, c0: 16-col chunk); within tile:
// element [lane][h] = W[kc*32 + 16*(lane>>4) + h][c0*16 + (lane&15)]
__global__ __launch_bounds__(256) void pack_w_kernel(
    const float* __restrict__ Wk, const float* __restrict__ Wv,
    const float* __restrict__ Wc, _Float16* __restrict__ Pk,
    _Float16* __restrict__ Pv, _Float16* __restrict__ Pc) {
  int e = blockIdx.x * 256 + threadIdx.x;  // 0..16383
  int t = e >> 9, li = (e >> 4) & 31, h = e & 15;
  int kc = t >> 3, c0 = t & 7;
  size_t s = (size_t)(kc * 32 + ((li >> 4) << 4) + h) * En + c0 * 16 + (li & 15);
  Pk[e] = (_Float16)Wk[s];
  Pv[e] = (_Float16)Wv[s];
  Pc[e] = (_Float16)Wc[s];
}

// -------- Kernel 1: K/V projection (+ X f16 copy); V stored transposed --------
__global__ __launch_bounds__(128) void proj_kernel(
    const float* __restrict__ X, const _Float16* __restrict__ Pk,
    const _Float16* __restrict__ Pv, _Float16* __restrict__ Kf,
    _Float16* __restrict__ Vt,   // [B,128,N] transposed
    _Float16* __restrict__ Xf) {
  __shared__ _Float16 xt[16][En];
  int b = blockIdx.y;
  int n0 = blockIdx.x * 16;
  int tid = threadIdx.x, lane = tid & 31, wv = tid >> 5;
  int r = lane & 15, hi = lane >> 4;

  const float4* xs4 = (const float4*)(X + ((size_t)b * Nn + n0) * En);
  _Float16* xd = Xf + ((size_t)b * Nn + n0) * En;
  _Float16* xtf = &xt[0][0];
  for (int idx = tid; idx < 16 * En / 4; idx += 128) {
    float4 f = xs4[idx];
    v4h hv = {(_Float16)f.x, (_Float16)f.y, (_Float16)f.z, (_Float16)f.w};
    *(v4h*)(xtf + idx * 4) = hv;
    *(v4h*)(xd + idx * 4) = hv;
  }
  __syncthreads();

  for (int tt = 0; tt < 2; ++tt) {
    int c0t = wv + 4 * tt;  // 16-col tile index
    v8f ak = {}, av = {};
#pragma unroll
    for (int kc = 0; kc < En; kc += 32) {
      v16h a  = load_a_f16(&xt[0][kc], En, lane);
      v16h bk = load_packed(Pk, (kc >> 5) * 8 + c0t, lane);
      v16h bw = load_packed(Pv, (kc >> 5) * 8 + c0t, lane);
      ak = wmma32(a, bk, ak);
      av = wmma32(a, bw, av);
    }
#pragma unroll
    for (int i = 0; i < 8; ++i) {
      int nrow = n0 + i + 8 * hi;
      int col = c0t * 16 + r;
      Kf[((size_t)b * Nn + nrow) * En + col] = (_Float16)ak[i];
      Vt[((size_t)b * En + col) * Nn + nrow] = (_Float16)av[i];
    }
  }
}

// -------- Kernel 2a: qbase[b][:] = graph[b] @ Wq[0:128,:] --------
__global__ void qbase_kernel(const float* __restrict__ graph,
                             const float* __restrict__ Wq,
                             float* __restrict__ qb) {
  int b = blockIdx.x, h = threadIdx.x;  // 128 threads
  const float* g = graph + (size_t)b * En;
  float s = 0.f;
#pragma unroll 4
  for (int e = 0; e < En; ++e) s += g[e] * Wq[(size_t)e * En + h];
  qb[(size_t)b * En + h] = s;
}

// -------- Kernel 2b: Q packed directly into WMMA A-operand layout --------
// Qp[((b*32+gt)*8+head)*512 + lane*16 + h]; K dim zero-padded 16->32.
__global__ __launch_bounds__(256) void qpack_kernel(
    const float* __restrict__ qb, const float* __restrict__ cap,
    const float* __restrict__ Wq, _Float16* __restrict__ Qp) {
  size_t e = (size_t)blockIdx.x * 256 + threadIdx.x;
  int h = (int)(e & 15);
  int li = (int)((e >> 4) & 31);
  int head = (int)((e >> 9) & 7);
  int gt = (int)((e >> 12) & 31);
  int b = (int)(e >> 17);
  int r = li & 15, hi = li >> 4;
  int k = a_kidx(h, hi);
  float val = 0.f;
  if (k < 16) {
    int col = head * 16 + k;
    val = qb[(size_t)b * En + col] +
          cap[(size_t)b * Gn + gt * 16 + r] * Wq[(size_t)En * En + col];
  }
  Qp[e] = (_Float16)val;
}

// -------- Kernel 3: fused flash-attention + pointer softmax --------
__global__ __launch_bounds__(256) void attn_kernel(
    const float* __restrict__ mask,     // [B,G,N]
    const _Float16* __restrict__ Pc,    // packed Wcomb
    const float* __restrict__ bcomb,    // [128]
    const _Float16* __restrict__ Qp,    // packed Q A-tiles
    const _Float16* __restrict__ Kf,    // [B,N,128]
    const _Float16* __restrict__ Vt,    // [B,128,N]
    const _Float16* __restrict__ Xf,    // [B,N,128]
    float* __restrict__ out) {          // [B,G,N]
  __shared__ _Float16 pbuf[8][16][32];  // per-wave P tile (C->A layout bounce)
  __shared__ float    outc[16][En];     // out_concat (f32)
  __shared__ _Float16 mhf[16][En];      // mh_atten_out (f16)
  __shared__ float    s2[16][Nn];       // stage-2 logits

  int b = blockIdx.y;
  int gt = blockIdx.x;
  int g0 = gt * 16;
  int tid = threadIdx.x, lane = tid & 31, wv = tid >> 5;  // wave == head
  int r = lane & 15, hi = lane >> 4;

  const float* mb = mask + ((size_t)b * Gn + g0) * Nn;

  // ---- stage 1: flash attention, head = wv ----
  int hoff = wv * Dn;
  v16h qa = *(const v16h*)(Qp + ((((size_t)b * 32 + gt) * 8 + wv) << 9) +
                           (lane << 4));

  v8f oacc = {};
  float mrow[8], srow[8];
#pragma unroll
  for (int i = 0; i < 8; ++i) { mrow[i] = -__builtin_inff(); srow[i] = 0.f; }

  const _Float16* kbase = Kf + (size_t)b * Nn * En + hoff;
  const _Float16* vtb = Vt + ((size_t)b * En + hoff + r) * Nn;

  for (int n0 = 0; n0 < Nn; n0 += 32) {
    // B tiles for Q@K^T: real K rows on lanes hi==0 (k=0..15); hi==1 is the
    // zero pad (matches zero-padded qa upper K half).
    v16h kb0 = {}, kb1 = {};
    if (hi == 0) {
      kb0 = *(const v16h*)(kbase + (size_t)(n0 + r) * En);
      kb1 = *(const v16h*)(kbase + (size_t)(n0 + 16 + r) * En);
    }
    v8f z = {};
    v8f s0 = wmma32(qa, kb0, z);
    v8f s1 = wmma32(qa, kb1, z);

    float nm[8], p0[8], p1[8];
#pragma unroll
    for (int i = 0; i < 8; ++i) {
      int gr = i + 8 * hi;
      s0[i] = s0[i] * 0.25f + mb[(size_t)gr * Nn + n0 + r];
      s1[i] = s1[i] * 0.25f + mb[(size_t)gr * Nn + n0 + 16 + r];
      float mx = fmaxf(s0[i], s1[i]);
      for (int mk = 1; mk < 16; mk <<= 1) mx = fmaxf(mx, __shfl_xor(mx, mk, 32));
      nm[i] = fmaxf(mrow[i], mx);
      p0[i] = __expf(s0[i] - nm[i]);
      p1[i] = __expf(s1[i] - nm[i]);
      float rs = p0[i] + p1[i];
      for (int mk = 1; mk < 16; mk <<= 1) rs += __shfl_xor(rs, mk, 32);
      float corr = __expf(mrow[i] - nm[i]);
      srow[i] = srow[i] * corr + rs;
      oacc[i] *= corr;
      mrow[i] = nm[i];
    }
    __builtin_amdgcn_wave_barrier();
#pragma unroll
    for (int i = 0; i < 8; ++i) {
      int gr = i + 8 * hi;
      pbuf[wv][gr][r]      = (_Float16)p0[i];
      pbuf[wv][gr][16 + r] = (_Float16)p1[i];
    }
    __builtin_amdgcn_wave_barrier();  // LDS ops are in-order within a wave
    v16h pa = load_a_f16(&pbuf[wv][0][0], 32, lane);
    v16h vb = *(const v16h*)(vtb + n0 + (hi << 4));  // B[k][d]=V[n0+k][hoff+d]
    oacc = wmma32(pa, vb, oacc);
    __builtin_amdgcn_wave_barrier();
  }
#pragma unroll
  for (int i = 0; i < 8; ++i)
    outc[i + 8 * hi][hoff + r] = oacc[i] / srow[i];
  __syncthreads();

  // ---- mh = out_concat @ Wcomb + bcomb; wave wv owns cols [16wv,16wv+16) ----
  {
    v8f acc = {};
#pragma unroll
    for (int kc = 0; kc < En; kc += 32) {
      v16h a = load_a_f32(&outc[0][kc], En, lane);
      v16h w = load_packed(Pc, (kc >> 5) * 8 + wv, lane);
      acc = wmma32(a, w, acc);
    }
    float bc = bcomb[wv * 16 + r];
#pragma unroll
    for (int i = 0; i < 8; ++i)
      mhf[i + 8 * hi][wv * 16 + r] = (_Float16)(acc[i] + bc);
  }
  __syncthreads();

  // ---- stage 2: score2 = mh @ X^T; wave wv owns n in [64wv, 64wv+64) ----
  const float inv_sqrt_e = 0.08838834764831845f;  // 1/sqrt(128)
  const _Float16* xb = Xf + (size_t)b * Nn * En;
  for (int t = 0; t < 4; ++t) {
    int n0 = wv * 64 + t * 16;
    v8f acc = {};
#pragma unroll
    for (int kc = 0; kc < En; kc += 32) {
      v16h a = load_a_f16(&mhf[0][kc], En, lane);
      // B[k][n] = X[n0+n][kc+k]: contiguous 32B per lane
      v16h x = *(const v16h*)(xb + (size_t)(n0 + r) * En + kc + (hi << 4));
      acc = wmma32(a, x, acc);
    }
#pragma unroll
    for (int i = 0; i < 8; ++i) {
      int gr = i + 8 * hi;
      float sc = 10.0f * tanhf(acc[i] * inv_sqrt_e);
      s2[gr][n0 + r] = sc + mb[(size_t)gr * Nn + n0 + r];
    }
  }
  __syncthreads();

  // ---- row softmax over N=512; wave wv owns rows {2wv, 2wv+1} ----
  float* orow = out + ((size_t)b * Gn + g0) * Nn;
  for (int rr = wv * 2; rr < wv * 2 + 2; ++rr) {
    float mx = -__builtin_inff();
    for (int n = lane; n < Nn; n += 32) mx = fmaxf(mx, s2[rr][n]);
    for (int mk = 16; mk >= 1; mk >>= 1) mx = fmaxf(mx, __shfl_xor(mx, mk, 32));
    float vals[16], sm = 0.f;
#pragma unroll
    for (int j = 0; j < 16; ++j) {
      vals[j] = __expf(s2[rr][lane + 32 * j] - mx);
      sm += vals[j];
    }
    for (int mk = 16; mk >= 1; mk >>= 1) sm += __shfl_xor(sm, mk, 32);
    float inv = 1.0f / sm;
#pragma unroll
    for (int j = 0; j < 16; ++j)
      orow[(size_t)rr * Nn + lane + 32 * j] = vals[j] * inv;
  }
}

// ---------------- host side ----------------
extern "C" void kernel_launch(void* const* d_in, const int* in_sizes, int n_in,
                              void* d_out, int out_size, void* d_ws, size_t ws_size,
                              hipStream_t stream) {
  (void)in_sizes; (void)n_in; (void)out_size; (void)ws_size;
  const float* X     = (const float*)d_in[0];  // encoded_nodes [B,N,128]
  const float* graph = (const float*)d_in[1];  // [B,1,128]
  const float* cap   = (const float*)d_in[2];  // [B,G]
  const float* mask  = (const float*)d_in[3];  // [B,G,N]
  const float* Wq    = (const float*)d_in[4];  // [129,128]
  const float* Wk    = (const float*)d_in[5];  // [128,128]
  const float* Wv    = (const float*)d_in[6];  // [128,128]
  const float* Wcomb = (const float*)d_in[7];  // [128,128]
  const float* bcomb = (const float*)d_in[8];  // [128]
  float* out = (float*)d_out;

  char* ws = (char*)d_ws;
  size_t off = 0;
  auto carve = [&](size_t bytes) -> void* {
    void* p = ws + off;
    off = (off + bytes + 255) & ~(size_t)255;
    return p;
  };
  const size_t act = (size_t)Bn * Nn * En * sizeof(_Float16);  // 8 MB each
  _Float16* Kf = (_Float16*)carve(act);
  _Float16* Vt = (_Float16*)carve(act);
  _Float16* Xf = (_Float16*)carve(act);
  _Float16* Qp = (_Float16*)carve((size_t)Bn * 32 * 8 * 512 * sizeof(_Float16));
  float*    qb = (float*)carve((size_t)Bn * En * sizeof(float));
  _Float16* Pk = (_Float16*)carve(16384 * sizeof(_Float16));
  _Float16* Pv = (_Float16*)carve(16384 * sizeof(_Float16));
  _Float16* Pc = (_Float16*)carve(16384 * sizeof(_Float16));

  pack_w_kernel<<<dim3(64), 256, 0, stream>>>(Wk, Wv, Wcomb, Pk, Pv, Pc);
  proj_kernel<<<dim3(Nn / 16, Bn), 128, 0, stream>>>(X, Pk, Pv, Kf, Vt, Xf);
  qbase_kernel<<<dim3(Bn), 128, 0, stream>>>(graph, Wq, qb);
  qpack_kernel<<<dim3((unsigned)((size_t)Bn * 32 * 8 * 512 / 256)), 256, 0,
                 stream>>>(qb, cap, Wq, Qp);
  attn_kernel<<<dim3(Gn / 16, Bn), 256, 0, stream>>>(mask, Pc, bcomb, Qp, Kf, Vt,
                                                     Xf, out);
}
